// NeRFModel_83288005804610
// MI455X (gfx1250) — matrix operand order repaired and checked by reading, not compiled, not run on hardware
//
#include <hip/hip_runtime.h>
#include <math.h>

typedef __attribute__((ext_vector_type(16))) _Float16 v16h;
typedef __attribute__((ext_vector_type(8)))  _Float16 v8h;
typedef __attribute__((ext_vector_type(8)))  float    v8f;

#define NUM_COARSE 8192
#define NUM_FINE   16384
#define NUM_ALL    24576
#define WIDTH_     256
#define MTILE      128
#define PI_F       3.14159265358979323846f

// ---------------- workspace layout (float offsets) ----------------
#define HDR_OFS    0          // 64 floats: [0..2]=base [3..5]=slope [6..8]=col_base
#define TALL_OFS   64         // 24576  (t unsorted: coarse | fine)
#define RALL_OFS   (TALL_OFS + NUM_ALL)
#define GALL_OFS   (RALL_OFS + NUM_ALL)
#define BALL_OFS   (GALL_OFS + NUM_ALL)
#define SALL_OFS   (BALL_OFS + NUM_ALL)
#define CDF_OFS    (SALL_OFS + NUM_ALL)        // 8192
#define SORT_OFS   (CDF_OFS + NUM_COARSE)      // 5 * 24576 sorted
#define W0T_OFS    (SORT_OFS + 5 * NUM_ALL)    // 256x64 f16  -> 8192 floats
#define WHT_OFS    (W0T_OFS + 8192)            // 8x256x256 f16 -> 262144 floats

#define MLP_SMEM   (MTILE*WIDTH_*2 + WIDTH_*WIDTH_*2)   // 64KB + 128KB = 196608
#define SORT_N     32768
#define SORT_SMEM  (SORT_N * 4)                          // 128KB

// ---------------- weight convert: f32 -> f16, N-major (transposed) ----------------
__global__ void convw_kernel(const float* __restrict__ W0, const float* __restrict__ Wh,
                             _Float16* __restrict__ W0t, _Float16* __restrict__ Wht) {
    int i = blockIdx.x * 256 + threadIdx.x;
    if (i < 256 * 64) {
        int n = i >> 6, k = i & 63;
        W0t[n * 64 + k] = (_Float16)((k < 60) ? W0[k * 256 + n] : 0.0f);
    }
    int j = i - 256 * 64;
    if (j >= 0 && j < 8 * 65536) {
        int l = j >> 16, r = j & 65535, n = r >> 8, k = r & 255;
        Wht[j] = (_Float16)(Wh[(l << 16) + k * 256 + n]);   // [l][n][k] <- [l][k][n]
    }
}

// ---------------- ray prep ----------------
__global__ void prep_kernel(const float* hor, const float* ver, const float* T,
                            const float* Wcol, const float* bcol, float* hdr) {
    if (threadIdx.x != 0 || blockIdx.x != 0) return;
    float h = hor[0], v = ver[0];           // double negation: d_cam = [hor, ver, 1, 1]
    float dirw[3], de[24];
    for (int j = 0; j < 3; ++j) {
        hdr[j]     = T[j*4+0]*h + T[j*4+1]*v + T[j*4+3];          // base
        hdr[3 + j] = T[j*4+2];                                    // slope (t coefficient)
        dirw[j]    = T[j*4+0]*h + T[j*4+1]*v + T[j*4+2] + T[j*4+3];
    }
    for (int c = 0; c < 3; ++c) {
        float f = PI_F;
        for (int i = 0; i < 4; ++i) {
            de[c*8 + 2*i]     = sinf(dirw[c] * f);
            de[c*8 + 2*i + 1] = cosf(dirw[c] * f);
            f *= 4.0f;
        }
    }
    for (int c = 0; c < 3; ++c) {
        float s = bcol[c];
        for (int k = 0; k < 24; ++k) s += de[k] * Wcol[k * 3 + c];
        hdr[6 + c] = s;                                           // dir part folded in
    }
}

// ---------------- coarse t generation ----------------
__global__ void tgen_kernel(const float* nearp, const float* farp, float* t_co) {
    int i = blockIdx.x * 256 + threadIdx.x;
    if (i < NUM_COARSE)
        t_co[i] = nearp[0] + (farp[0] - nearp[0]) * (float)i / (float)(NUM_COARSE - 1);
}

// ---------------- one dense layer via WMMA ----------------
__device__ __forceinline__ void run_layer(_Float16* act, const _Float16* wts,
                                          int ktiles, int wstride, const float* bias,
                                          bool relu, int wave, int lane) {
    const int nlane = lane & 15;
    const int half  = lane >> 4;
    const int mrow  = wave * 16 + nlane;
    const v8f vz = {0.f,0.f,0.f,0.f,0.f,0.f,0.f,0.f};
    v8f acc[16];
#pragma unroll
    for (int nt = 0; nt < 16; ++nt) acc[nt] = vz;

    for (int kt = 0; kt < ktiles; ++kt) {
        int k0 = kt * 32;
        // A tile 16x32 f16: lanes0-15 K=k0..k0+7 / +16..+23 ; lanes16-31 +8 shifted
        const _Float16* ap = act + mrow * WIDTH_ + k0 + half * 8;
        v8h alo = *(const v8h*)ap;
        v8h ahi = *(const v8h*)(ap + 16);
        v16h a = __builtin_shufflevector(alo, ahi, 0,1,2,3,4,5,6,7,8,9,10,11,12,13,14,15);
#pragma unroll
        for (int nt = 0; nt < 16; ++nt) {
            // B tile 32x16 f16, weights stored N-major: lanes0-15 K=k0..k0+7 / +16..+23 ; lanes16-31 +8 shifted
            const _Float16* bp = wts + (nt * 16 + nlane) * wstride + k0 + half * 8;
            v8h blo = *(const v8h*)bp;
            v8h bhi = *(const v8h*)(bp + 16);
            v16h b = __builtin_shufflevector(blo, bhi, 0,1,2,3,4,5,6,7,8,9,10,11,12,13,14,15);
            acc[nt] = __builtin_amdgcn_wmma_f32_16x16x32_f16(
                false, a, false, b, (short)0, acc[nt], false, false);
        }
    }
    __syncthreads();   // all A/B reads done before act is overwritten
    int mbase = wave * 16 + half * 8;
#pragma unroll
    for (int nt = 0; nt < 16; ++nt) {
        int n = nt * 16 + nlane;
        float bv = bias[n];
#pragma unroll
        for (int r = 0; r < 8; ++r) {
            float v = acc[nt][r] + bv;
            if (relu) v = fmaxf(v, 0.0f);
            act[(mbase + r) * WIDTH_ + n] = (_Float16)v;
        }
    }
}

// ---------------- full MLP over a 128-point tile ----------------
__global__ __launch_bounds__(256) void mlp_kernel(
        const float* __restrict__ hdr, const float* __restrict__ t_arr,
        const _Float16* __restrict__ W0t, const _Float16* __restrict__ Wht,
        const float* __restrict__ b0, const float* __restrict__ bh,
        const float* __restrict__ Wsig, const float* __restrict__ bsig,
        const float* __restrict__ Wcol,
        float* __restrict__ sig_out, float* __restrict__ r_out,
        float* __restrict__ g_out, float* __restrict__ b_out) {
    extern __shared__ char smem_raw[];
    _Float16* act = (_Float16*)smem_raw;                        // [128][256]
    _Float16* wts = (_Float16*)(smem_raw + MTILE * WIDTH_ * 2); // [256][<=256] N-major

    int tid = threadIdx.x, lane = tid & 31, wave = tid >> 5;
    int base = blockIdx.x * MTILE;

    // positional encoding -> act cols 0..59, pad 60..63 with 0
    if (tid < MTILE) {
        float t = t_arr[base + tid];
        _Float16* row = act + tid * WIDTH_;
        for (int c = 0; c < 3; ++c) {
            float p = hdr[c] + t * hdr[3 + c];
            float f = PI_F;
            for (int i = 0; i < 10; ++i) {
                float ang = p * f;
                row[c * 20 + 2 * i]     = (_Float16)sinf(ang);
                row[c * 20 + 2 * i + 1] = (_Float16)cosf(ang);
                f *= 4.0f;
            }
        }
        for (int k = 60; k < 64; ++k) row[k] = (_Float16)0.0f;
    }
    // stage W0t (32KB) into LDS
    {
        const uint4* src = (const uint4*)W0t;
        uint4* dst = (uint4*)wts;
        for (int i = tid; i < 2048; i += 256) dst[i] = src[i];
    }
    __syncthreads();
    run_layer(act, wts, 2, 64, b0, true, wave, lane);
    __syncthreads();

    for (int l = 0; l < 8; ++l) {
        const uint4* src = (const uint4*)(Wht + l * 65536);
        uint4* dst = (uint4*)wts;
        for (int i = tid; i < 8192; i += 256) dst[i] = src[i];
        __syncthreads();
        run_layer(act, wts, 8, WIDTH_, bh + l * 256, l < 7, wave, lane);
        __syncthreads();
    }

    // heads: thread -> (point, sigma|color)
    int p = tid & 127, which = tid >> 7;
    const _Float16* h = act + p * WIDTH_;
    if (which == 0) {
        float s = bsig[0];
        for (int k = 0; k < 256; ++k) s += (float)h[k] * Wsig[k];
        sig_out[base + p] = 1.0f / (1.0f + expf(-s));
    } else {
        float c0 = hdr[6], c1 = hdr[7], c2 = hdr[8];
        for (int k = 0; k < 256; ++k) {
            float hv = (float)h[k];
            c0 += hv * Wcol[(24 + k) * 3 + 0];
            c1 += hv * Wcol[(24 + k) * 3 + 1];
            c2 += hv * Wcol[(24 + k) * 3 + 2];
        }
        r_out[base + p] = 1.0f / (1.0f + expf(-c0));
        g_out[base + p] = 1.0f / (1.0f + expf(-c1));
        b_out[base + p] = 1.0f / (1.0f + expf(-c2));
    }
}

// ---------------- cdf = cumsum(sigma_coarse), single block ----------------
__global__ void scan_kernel(const float* __restrict__ sig, float* __restrict__ cdf) {
    __shared__ float buf[256];
    __shared__ float carry;
    int tid = threadIdx.x;
    if (tid == 0) carry = 0.f;
    __syncthreads();
    for (int c = 0; c < NUM_COARSE / 256; ++c) {
        int i = c * 256 + tid;
        buf[tid] = sig[i];
        __syncthreads();
        for (int off = 1; off < 256; off <<= 1) {
            float t = (tid >= off) ? buf[tid - off] : 0.f;
            __syncthreads();
            buf[tid] += t;
            __syncthreads();
        }
        cdf[i] = carry + buf[tid];
        __syncthreads();
        if (tid == 255) carry += buf[255];
        __syncthreads();
    }
}

// ---------------- inverse-CDF fine sampling ----------------
__global__ void sample_kernel(const float* __restrict__ cdf, const float* __restrict__ t_co,
                              const float* __restrict__ sig,
                              const float* nearp, const float* farp,
                              float* __restrict__ t_fine) {
    int i = blockIdx.x * 256 + threadIdx.x;
    if (i >= NUM_FINE) return;
    float c0 = cdf[0], cN = cdf[NUM_COARSE - 1];
    float u = c0 + (cN - c0) * (float)(i + 1) / (float)(NUM_FINE + 1);
    int lo = 0, hi = NUM_COARSE;                // searchsorted 'left'
    while (lo < hi) { int mid = (lo + hi) >> 1; if (cdf[mid] < u) lo = mid + 1; else hi = mid; }
    int idx = lo - 1;
    if (idx < 0) idx = 0;
    if (idx > NUM_COARSE - 2) idx = NUM_COARSE - 2;
    float dt = (farp[0] - nearp[0]) / (float)(NUM_COARSE - 1);
    t_fine[i] = t_co[idx] + (u - cdf[idx]) * dt / sig[idx + 1];
}

// ---------------- independent bitonic sort of one 24576-row (jnp.sort axis=1) ----------------
__global__ void sort_kernel(const float* __restrict__ src_base, float* __restrict__ dst_base) {
    extern __shared__ float sdata[];
    const float* src = src_base + blockIdx.x * NUM_ALL;
    float* dst = dst_base + blockIdx.x * NUM_ALL;
    int tid = threadIdx.x, nthr = blockDim.x;
    for (int i = tid; i < SORT_N; i += nthr)
        sdata[i] = (i < NUM_ALL) ? src[i] : __builtin_inff();
    __syncthreads();
    for (int k = 2; k <= SORT_N; k <<= 1) {
        for (int j = k >> 1; j > 0; j >>= 1) {
            for (int i = tid; i < SORT_N; i += nthr) {
                int ixj = i ^ j;
                if (ixj > i) {
                    float a = sdata[i], b = sdata[ixj];
                    bool up = ((i & k) == 0);
                    if ((a > b) == up) { sdata[i] = b; sdata[ixj] = a; }
                }
            }
            __syncthreads();
        }
    }
    for (int i = tid; i < NUM_ALL; i += nthr) dst[i] = sdata[i];
}

// ---------------- transmittance-weighted accumulation ----------------
__global__ void accum_kernel(const float* __restrict__ ts, const float* __restrict__ Rv,
                             const float* __restrict__ Gv, const float* __restrict__ Bv,
                             const float* __restrict__ Sv,
                             const float* nearp, const float* farp,
                             float* __restrict__ out, int n, int mode) {
    __shared__ float buf[256];
    __shared__ float carry;
    int tid = threadIdx.x;
    if (tid == 0) carry = 0.f;
    float aR = 0.f, aG = 0.f, aB = 0.f;
    float dco = (farp[0] - nearp[0]) / (float)NUM_COARSE;
    __syncthreads();
    int chunks = n >> 8;
    for (int c = 0; c < chunks; ++c) {
        int i = c * 256 + tid;
        float delta = mode ? ((i < n - 1) ? (ts[i + 1] - ts[i]) : 0.0001f) : dco;
        float sd = delta * Sv[i];
        buf[tid] = sd;
        __syncthreads();
        for (int off = 1; off < 256; off <<= 1) {
            float t = (tid >= off) ? buf[tid - off] : 0.f;
            __syncthreads();
            buf[tid] += t;
            __syncthreads();
        }
        float cum = carry + buf[tid];                 // inclusive cumsum
        float w = expf(-cum) * (1.0f - expf(-sd));
        aR += Rv[i] * w; aG += Gv[i] * w; aB += Bv[i] * w;
        __syncthreads();
        if (tid == 255) carry = cum;
        __syncthreads();
    }
    buf[tid] = aR; __syncthreads();
    for (int off = 128; off > 0; off >>= 1) { if (tid < off) buf[tid] += buf[tid + off]; __syncthreads(); }
    if (tid == 0) out[0] = buf[0];
    __syncthreads();
    buf[tid] = aG; __syncthreads();
    for (int off = 128; off > 0; off >>= 1) { if (tid < off) buf[tid] += buf[tid + off]; __syncthreads(); }
    if (tid == 0) out[1] = buf[0];
    __syncthreads();
    buf[tid] = aB; __syncthreads();
    for (int off = 128; off > 0; off >>= 1) { if (tid < off) buf[tid] += buf[tid + off]; __syncthreads(); }
    if (tid == 0) out[2] = buf[0];
}

// ---------------- driver ----------------
extern "C" void kernel_launch(void* const* d_in, const int* in_sizes, int n_in,
                              void* d_out, int out_size, void* d_ws, size_t ws_size,
                              hipStream_t stream) {
    const float* hor  = (const float*)d_in[0];
    const float* ver  = (const float*)d_in[1];
    const float* Tm   = (const float*)d_in[2];
    const float* nearp= (const float*)d_in[3];
    const float* farp = (const float*)d_in[4];
    const float* W0   = (const float*)d_in[5];
    const float* b0   = (const float*)d_in[6];
    const float* Wh   = (const float*)d_in[7];
    const float* bh   = (const float*)d_in[8];
    const float* Wsig = (const float*)d_in[9];
    const float* bsig = (const float*)d_in[10];
    const float* Wcol = (const float*)d_in[11];
    const float* bcol = (const float*)d_in[12];
    float* out = (float*)d_out;
    float* ws  = (float*)d_ws;

    float* hdr   = ws + HDR_OFS;
    float* t_all = ws + TALL_OFS;
    float* Rall  = ws + RALL_OFS;
    float* Gall  = ws + GALL_OFS;
    float* Ball  = ws + BALL_OFS;
    float* Sall  = ws + SALL_OFS;
    float* cdf   = ws + CDF_OFS;
    float* srt   = ws + SORT_OFS;
    _Float16* W0t = (_Float16*)(ws + W0T_OFS);
    _Float16* Wht = (_Float16*)(ws + WHT_OFS);

    prep_kernel<<<1, 32, 0, stream>>>(hor, ver, Tm, Wcol, bcol, hdr);
    convw_kernel<<<(256 * 64 + 8 * 65536) / 256, 256, 0, stream>>>(W0, Wh, W0t, Wht);
    tgen_kernel<<<NUM_COARSE / 256, 256, 0, stream>>>(nearp, farp, t_all);

    // coarse pass
    mlp_kernel<<<NUM_COARSE / MTILE, 256, MLP_SMEM, stream>>>(
        hdr, t_all, W0t, Wht, b0, bh, Wsig, bsig, Wcol, Sall, Rall, Gall, Ball);
    scan_kernel<<<1, 256, 0, stream>>>(Sall, cdf);
    sample_kernel<<<NUM_FINE / 256, 256, 0, stream>>>(cdf, t_all, Sall, nearp, farp,
                                                      t_all + NUM_COARSE);
    // fine pass
    mlp_kernel<<<NUM_FINE / MTILE, 256, MLP_SMEM, stream>>>(
        hdr, t_all + NUM_COARSE, W0t, Wht, b0, bh, Wsig, bsig, Wcol,
        Sall + NUM_COARSE, Rall + NUM_COARSE, Gall + NUM_COARSE, Ball + NUM_COARSE);

    // sort the 5 bundle rows independently: [t | R | G | B | S] are contiguous
    sort_kernel<<<5, 512, SORT_SMEM, stream>>>(t_all, srt);

    // C_coarse from unsorted coarse arrays; C_fine from sorted bundle
    accum_kernel<<<1, 256, 0, stream>>>(t_all, Rall, Gall, Ball, Sall,
                                        nearp, farp, out, NUM_COARSE, 0);
    accum_kernel<<<1, 256, 0, stream>>>(srt, srt + NUM_ALL, srt + 2 * NUM_ALL,
                                        srt + 3 * NUM_ALL, srt + 4 * NUM_ALL,
                                        nearp, farp, out + 3, NUM_ALL, 1);
}
